// BiLSTMCRF_74603581932106
// MI455X (gfx1250) — matrix operand order repaired and checked by reading, not compiled, not run on hardware
//
#include <hip/hip_runtime.h>
#include <hip/hip_bf16.h>

typedef __attribute__((ext_vector_type(16))) __bf16 v16bf;
typedef __attribute__((ext_vector_type(8)))  __bf16 v8bf;
typedef __attribute__((ext_vector_type(8)))  float  v8f;

#define H300 300
#define HP   320      // padded hidden (K for recurrent GEMM)
#define G4H  1200     // 4*H gate width (gate-major reordered)
#define BB   64
#define TT   512
#define LL   34
#define LP   48       // padded label count
#define K2P  640      // padded 2*H (layer-2 input K)
#define NTW  5        // N-tiles per wave in the big GEMM (75 = 15 * 5)

// ---------------------------------------------------------------- WMMA utils
__device__ __forceinline__ v8f wmma_bf16(v16bf a, v16bf b, v8f c) {
  // v_wmma_f32_16x16x32_bf16: (neg_a, A, neg_b, B, c_mod, C, reuse_a, reuse_b)
  return __builtin_amdgcn_wmma_f32_16x16x32_bf16(false, a, false, b, (short)0, c,
                                                 false, false);
}

// A fragment (16x32 bf16): lane L<16 -> M=L, K = {0..15}; lane L+16 ->
// M=L, K = {16..31}.  With p = rowBase + khalf*16, the two 8-element
// groups are contiguous 16B loads.
__device__ __forceinline__ v16bf load_a16(const __bf16* p) {
  v8bf lo = *(const v8bf*)(p);
  v8bf hi = *(const v8bf*)(p + 8);
  return __builtin_shufflevector(lo, hi, 0,1,2,3,4,5,6,7,8,9,10,11,12,13,14,15);
}

__device__ __forceinline__ float sigf(float x) { return 1.0f / (1.0f + __expf(-x)); }

// ---------------------------------------------------------------- zero fills
__global__ void zero_bf16_kernel(__bf16* p, size_t n) {
  size_t i = (size_t)blockIdx.x * blockDim.x + threadIdx.x;
  size_t stride = (size_t)gridDim.x * blockDim.x;
  for (; i < n; i += stride) p[i] = (__bf16)0.0f;
}
__global__ void zero_f32_kernel(float* p, size_t n) {
  size_t i = (size_t)blockIdx.x * blockDim.x + threadIdx.x;
  size_t stride = (size_t)gridDim.x * blockDim.x;
  for (; i < n; i += stride) p[i] = 0.0f;
}

// ---------------------------------------------------------------- embedding
// x1 is time-major (T, B, HP) bf16; pad cols >= 300 are zero.
__global__ void embed_kernel(const int* __restrict__ ids,
                             const float* __restrict__ emb,
                             __bf16* __restrict__ x1) {
  size_t i = (size_t)blockIdx.x * blockDim.x + threadIdx.x;
  size_t total = (size_t)TT * BB * HP;
  if (i >= total) return;
  int k = (int)(i % HP);
  size_t row = i / HP;
  int t = (int)(row / BB);
  int b = (int)(row % BB);
  float v = 0.0f;
  if (k < H300) {
    int id = ids[(size_t)b * TT + t];
    v = emb[(size_t)id * H300 + k];
  }
  x1[i] = (__bf16)v;
}

// ------------------------------------------------- weight swizzle (B-layout)
// B fragment (32x16 bf16): lane = n_local + 16*(k_local>=16), slot i = k_local&15.
// Tiles stored as (ntile*kTiles + kt)*512 + lane*16 + i  => one 32B load/lane.
// Gate reorder: dst col n' = 4*j + gate  <- src row r = gate*300 + j.
__global__ void prep_wb(const float* __restrict__ W, __bf16* __restrict__ dst,
                        int in_dim, int Kp) {
  size_t idx = (size_t)blockIdx.x * blockDim.x + threadIdx.x;
  size_t total = (size_t)G4H * Kp;
  if (idx >= total) return;
  int np = (int)(idx / Kp);   // 0..1199 (gate-major)
  int k  = (int)(idx % Kp);
  int j = np >> 2, gate = np & 3;
  float v = (k < in_dim) ? W[(size_t)(gate * H300 + j) * in_dim + k] : 0.0f;
  int kt = k >> 5, kl = k & 31;
  int ntile = np >> 4, nl = np & 15;
  int lane = nl + ((kl >> 4) << 4);
  int slot = kl & 15;
  int kTiles = Kp >> 5;
  dst[(((size_t)ntile * kTiles + kt) << 9) + lane * 16 + slot] = (__bf16)v;
}

// classifier weight: cls_W is already (K=600, N=34); pad to (640, 48), swizzle.
__global__ void prep_cls(const float* __restrict__ Wc, __bf16* __restrict__ dst) {
  size_t idx = (size_t)blockIdx.x * blockDim.x + threadIdx.x;
  size_t total = (size_t)K2P * LP;
  if (idx >= total) return;
  int np = (int)(idx / K2P);
  int k  = (int)(idx % K2P);
  float v = (k < 2 * H300 && np < LL) ? Wc[(size_t)k * LL + np] : 0.0f;
  int kt = k >> 5, kl = k & 31;
  int ntile = np >> 4, nl = np & 15;
  int lane = nl + ((kl >> 4) << 4);
  int slot = kl & 15;
  dst[(((size_t)ntile * 20 + kt) << 9) + lane * 16 + slot] = (__bf16)v;
}

__global__ void prep_bias(const float* __restrict__ bih,
                          const float* __restrict__ bhh,
                          float* __restrict__ dst) {
  int np = blockIdx.x * blockDim.x + threadIdx.x;
  if (np >= G4H) return;
  int j = np >> 2, gate = np & 3;
  dst[np] = bih[gate * H300 + j] + bhh[gate * H300 + j];
}

// --------------------------------------- big input-projection GEMM (M=32768)
// xproj[row, n'] = sum_k xin[row,k] * Wih_sw[k,n'] + bias[n'] ; row time-major.
// Each wave owns 1 M-tile x 5 N-tiles: one A fragment feeds 5 WMMA per K-step
// (5x reduction in activation traffic vs one-tile-per-wave).
__global__ __launch_bounds__(32) void gemm_inproj(
    const __bf16* __restrict__ xin, const __bf16* __restrict__ Bsw,
    const float* __restrict__ bias, float* __restrict__ xproj, int Kp) {
  int ng = blockIdx.x;          // 0..14 -> N-tiles ng*5 .. ng*5+4
  int mt = blockIdx.y;          // 0..2047
  int lane = threadIdx.x;
  int laneM = lane & 15, khalf = lane >> 4;
  int kTiles = Kp >> 5;
  const __bf16* arow = xin + (size_t)(mt * 16 + laneM) * Kp + khalf * 16;
  const __bf16* bbase = Bsw + lane * 16;
  v8f acc[NTW] = {};
  for (int kt = 0; kt < kTiles; ++kt) {
    v16bf a = load_a16(arow + kt * 32);
    if (kt + 1 < kTiles)   // keep next A chunk warm in the near cache
      __builtin_prefetch((const void*)(arow + (kt + 1) * 32), 0, 3);
#pragma unroll
    for (int q = 0; q < NTW; ++q) {
      int nt = ng * NTW + q;
      v16bf b = *(const v16bf*)(bbase + (((size_t)nt * kTiles + kt) << 9));
      acc[q] = wmma_bf16(a, b, acc[q]);
    }
  }
  int moff = (lane >= 16) ? 8 : 0;
#pragma unroll
  for (int q = 0; q < NTW; ++q) {
    int n = (ng * NTW + q) * 16 + (lane & 15);
    float bv = bias[n];
#pragma unroll
    for (int r = 0; r < 8; ++r) {
      int m = mt * 16 + r + moff;
      xproj[(size_t)m * G4H + n] = acc[q][r] + bv;
    }
  }
}

// --------------------------------------------------- fused LSTM step kernel
// g = hprev @ Whh^T (WMMA, K=320) + xproj[t] (bias pre-folded); gate math in
// the same wave via a 16x16 LDS tile (gate-major layout => 4 hidden units).
__global__ __launch_bounds__(32) void lstm_step(
    const __bf16* __restrict__ hprev, __bf16* __restrict__ hout,
    const __bf16* __restrict__ WhhSw, const float* __restrict__ xproj,
    float* __restrict__ cbuf, __bf16* __restrict__ lout, int t, int colbase) {
  __shared__ float tile[16][17];
  int nt = blockIdx.x;          // 0..74 -> hidden units nt*4 .. nt*4+3
  int mt = blockIdx.y;          // 0..3  -> batch rows mt*16 ..
  int lane = threadIdx.x;
  int laneM = lane & 15, khalf = lane >> 4;
  const __bf16* arow = hprev + (size_t)(mt * 16 + laneM) * HP + khalf * 16;
  const __bf16* bptr = WhhSw + (((size_t)nt * 10) << 9) + lane * 16;
  v8f acc = {};
#pragma unroll
  for (int kt = 0; kt < 10; ++kt) {
    v16bf a = load_a16(arow + kt * 32);
    v16bf b = *(const v16bf*)(bptr + ((size_t)kt << 9));
    acc = wmma_bf16(a, b, acc);
  }
  int n = nt * 16 + (lane & 15);
  int moff = (lane >= 16) ? 8 : 0;
  const float* xq = xproj + (size_t)t * BB * G4H;
#pragma unroll
  for (int r = 0; r < 8; ++r) {
    int m = mt * 16 + r + moff;
    tile[r + moff][lane & 15] = acc[r] + xq[(size_t)m * G4H + n];
  }
  __syncthreads();
  // 64 (batch,hidden) cells in this tile; 2 per lane
  for (int cidx = lane; cidx < 64; cidx += 32) {
    int mloc = cidx >> 2, jl = cidx & 3;
    float gi = tile[mloc][jl * 4 + 0];
    float gf = tile[mloc][jl * 4 + 1];
    float gg = tile[mloc][jl * 4 + 2];
    float go = tile[mloc][jl * 4 + 3];
    int m = mt * 16 + mloc;
    int j = nt * 4 + jl;                    // 0..299
    size_t ci = (size_t)m * H300 + j;
    float c = cbuf[ci];
    float cn = sigf(gf) * c + sigf(gi) * tanhf(gg);
    float h = sigf(go) * tanhf(cn);
    cbuf[ci] = cn;
    hout[(size_t)m * HP + j] = (__bf16)h;
    lout[((size_t)t * BB + m) * (2 * HP) + colbase + j] = (__bf16)h;
  }
}

// ---------------------------------------------------------- classifier GEMM
// All 3 N-tiles (48 padded cols) in one wave: A loaded once per K-step.
__global__ __launch_bounds__(32) void gemm_cls(
    const __bf16* __restrict__ xin, const __bf16* __restrict__ Bsw,
    const float* __restrict__ bias, float* __restrict__ out) {
  int mt = blockIdx.x;          // 0..2047
  int lane = threadIdx.x;
  int laneM = lane & 15, khalf = lane >> 4;
  const __bf16* arow = xin + (size_t)(mt * 16 + laneM) * K2P + khalf * 16;
  const __bf16* bbase = Bsw + lane * 16;
  v8f acc[3] = {};
  for (int kt = 0; kt < 20; ++kt) {
    v16bf a = load_a16(arow + kt * 32);
#pragma unroll
    for (int q = 0; q < 3; ++q) {
      v16bf b = *(const v16bf*)(bbase + (((size_t)q * 20 + kt) << 9));
      acc[q] = wmma_bf16(a, b, acc[q]);
    }
  }
  int moff = (lane >= 16) ? 8 : 0;
#pragma unroll
  for (int q = 0; q < 3; ++q) {
    int n = q * 16 + (lane & 15);
    if (n < LL) {
      float bv = bias[n];
#pragma unroll
      for (int r = 0; r < 8; ++r) {
        int m = mt * 16 + r + moff;      // time-major row
        int t = m / BB, b = m % BB;
        out[((size_t)b * TT + t) * LL + n] = acc[q][r] + bv;
      }
    }
  }
}

// ------------------------------------------------------------------ CRF llh
__global__ __launch_bounds__(64) void crf_kernel(
    const float* __restrict__ logits, const int* __restrict__ mask,
    const int* __restrict__ labels, const float* __restrict__ startv,
    const float* __restrict__ endv, const float* __restrict__ trans,
    float* __restrict__ llh) {
  __shared__ float str[LL * LL];
  __shared__ float alpha[LL];
  __shared__ float redf[64];
  __shared__ int redi[64];
  int b = blockIdx.x;
  int tid = threadIdx.x;
  const float* em = logits + (size_t)b * TT * LL;
  const int* tg = labels + (size_t)b * TT;
  const int* mk = mask + (size_t)b * TT;
  for (int i = tid; i < LL * LL; i += 64) str[i] = trans[i];
  if (tid < LL) alpha[tid] = startv[tid] + em[tid];
  __syncthreads();
  for (int t = 1; t < TT; ++t) {
    float nv = 0.0f;
    if (tid < LL) {
      if (mk[t] != 0) {
        float mx = -3.402823466e+38f;
        for (int i = 0; i < LL; ++i) mx = fmaxf(mx, alpha[i] + str[i * LL + tid]);
        float s = 0.0f;
        for (int i = 0; i < LL; ++i) s += __expf(alpha[i] + str[i * LL + tid] - mx);
        nv = mx + __logf(s) + em[(size_t)t * LL + tid];
      } else {
        nv = alpha[tid];
      }
    }
    __syncthreads();
    if (tid < LL) alpha[tid] = nv;
    __syncthreads();
  }
  float part = 0.0f; int cnt = 0;
  for (int t = tid; t < TT; t += 64) {
    int m = mk[t];
    cnt += (m != 0);
    if (t >= 1 && m != 0)
      part += str[tg[t - 1] * LL + tg[t]] + em[(size_t)t * LL + tg[t]];
  }
  redf[tid] = part; redi[tid] = cnt;
  __syncthreads();
  if (tid == 0) {
    float mx = -3.402823466e+38f;
    for (int j = 0; j < LL; ++j) mx = fmaxf(mx, alpha[j] + endv[j]);
    float s = 0.0f;
    for (int j = 0; j < LL; ++j) s += __expf(alpha[j] + endv[j] - mx);
    float den = mx + __logf(s);
    float num = startv[tg[0]] + em[tg[0]];
    int total = 0;
    for (int i = 0; i < 64; ++i) { num += redf[i]; total += redi[i]; }
    num += endv[tg[total - 1]];
    llh[b] = num - den;
  }
}

__global__ __launch_bounds__(64) void loss_kernel(const float* __restrict__ llh,
                                                  float* __restrict__ out) {
  __shared__ float red[64];
  red[threadIdx.x] = llh[threadIdx.x];
  __syncthreads();
  if (threadIdx.x == 0) {
    float s = 0.0f;
    for (int i = 0; i < 64; ++i) s += red[i];
    out[0] = -s / 64.0f;
  }
}

// ------------------------------------------------------------------ launcher
extern "C" void kernel_launch(void* const* d_in, const int* in_sizes, int n_in,
                              void* d_out, int out_size, void* d_ws, size_t ws_size,
                              hipStream_t stream) {
  (void)in_sizes; (void)n_in; (void)out_size; (void)ws_size;
  // jax pytree leaf order (dict keys sorted): cls_W, cls_b, crf.{end,start,trans},
  // emb, lstm[0].{b,f}.{Whh,Wih,bhh,bih}, lstm[1].{b,f}.{...}
  const int*   input_ids = (const int*)d_in[0];
  const int*   attn      = (const int*)d_in[1];
  const int*   labels    = (const int*)d_in[2];
  const float* cls_W     = (const float*)d_in[3];
  const float* cls_b     = (const float*)d_in[4];
  const float* crf_end   = (const float*)d_in[5];
  const float* crf_start = (const float*)d_in[6];
  const float* crf_trans = (const float*)d_in[7];
  const float* embt      = (const float*)d_in[8];
  const float* W[4][4];  // [l0b, l0f, l1b, l1f][Whh, Wih, bhh, bih]
  for (int p = 0; p < 4; ++p)
    for (int q = 0; q < 4; ++q) W[p][q] = (const float*)d_in[9 + p * 4 + q];

  char* wsp = (char*)d_ws;
  auto alloc = [&](size_t bytes) {
    void* p = (void*)wsp;
    wsp += (bytes + 255) & ~(size_t)255;
    return p;
  };
  __bf16* x1    = (__bf16*)alloc((size_t)TT * BB * HP * 2);
  __bf16* l1out = (__bf16*)alloc((size_t)TT * BB * (2 * HP) * 2);
  __bf16* l2out = (__bf16*)alloc((size_t)TT * BB * (2 * HP) * 2);
  float*  xproj = (float*) alloc((size_t)TT * BB * G4H * 4);
  int kpIn[4] = {HP, HP, K2P, K2P};
  __bf16* wihSw[4]; __bf16* whhSw[4]; float* biasg[4];
  for (int p = 0; p < 4; ++p) {
    wihSw[p] = (__bf16*)alloc((size_t)kpIn[p] * G4H * 2);
    whhSw[p] = (__bf16*)alloc((size_t)HP * G4H * 2);
    biasg[p] = (float*) alloc((size_t)G4H * 4);
  }
  __bf16* clsSw = (__bf16*)alloc((size_t)K2P * LP * 2);
  __bf16* hbufA = (__bf16*)alloc((size_t)BB * HP * 2);
  __bf16* hbufB = (__bf16*)alloc((size_t)BB * HP * 2);
  float*  cbuf  = (float*) alloc((size_t)BB * H300 * 4);
  float*  llh   = (float*) alloc(64 * 4);

  // ---- prep: embed, zero layer outputs, swizzle weights
  {
    size_t n = (size_t)TT * BB * HP;
    embed_kernel<<<(unsigned)((n + 255) / 256), 256, 0, stream>>>(input_ids, embt, x1);
  }
  zero_bf16_kernel<<<4096, 256, 0, stream>>>(l1out, (size_t)TT * BB * 2 * HP);
  zero_bf16_kernel<<<4096, 256, 0, stream>>>(l2out, (size_t)TT * BB * 2 * HP);
  for (int p = 0; p < 4; ++p) {
    int in_dim = (p < 2) ? H300 : 2 * H300;
    size_t n1 = (size_t)G4H * kpIn[p];
    prep_wb<<<(unsigned)((n1 + 255) / 256), 256, 0, stream>>>(W[p][1], wihSw[p], in_dim, kpIn[p]);
    size_t n2 = (size_t)G4H * HP;
    prep_wb<<<(unsigned)((n2 + 255) / 256), 256, 0, stream>>>(W[p][0], whhSw[p], H300, HP);
    prep_bias<<<(G4H + 255) / 256, 256, 0, stream>>>(W[p][3], W[p][2], biasg[p]);
  }
  {
    size_t n = (size_t)K2P * LP;
    prep_cls<<<(unsigned)((n + 255) / 256), 256, 0, stream>>>(cls_W, clsSw);
  }

  // ---- 4 direction-layer passes (share xproj buffer sequentially)
  struct PassDesc { const __bf16* xin; int Kp; int widx; __bf16* lout; int colbase; bool fwd; };
  PassDesc passes[4] = {
    { x1,    HP,  1, l1out, 0,    true  },   // layer0 fwd
    { x1,    HP,  0, l1out, H300, false },   // layer0 bwd
    { l1out, K2P, 3, l2out, 0,    true  },   // layer1 fwd
    { l1out, K2P, 2, l2out, H300, false },   // layer1 bwd
  };
  for (int pi = 0; pi < 4; ++pi) {
    PassDesc& P = passes[pi];
    zero_bf16_kernel<<<64, 256, 0, stream>>>(hbufA, (size_t)BB * HP);
    zero_bf16_kernel<<<64, 256, 0, stream>>>(hbufB, (size_t)BB * HP);
    zero_f32_kernel<<<64, 256, 0, stream>>>(cbuf, (size_t)BB * H300);
    gemm_inproj<<<dim3((G4H / 16) / NTW, (TT * BB) / 16), 32, 0, stream>>>(
        P.xin, wihSw[P.widx], biasg[P.widx], xproj, P.Kp);
    __bf16* ha = hbufA; __bf16* hb = hbufB;
    for (int s = 0; s < TT; ++s) {
      int t = P.fwd ? s : (TT - 1 - s);
      lstm_step<<<dim3(G4H / 16, BB / 16), 32, 0, stream>>>(
          ha, hb, whhSw[P.widx], xproj, cbuf, P.lout, t, P.colbase);
      __bf16* tmp = ha; ha = hb; hb = tmp;
    }
  }

  // ---- classifier + CRF
  float* outf = (float*)d_out;
  gemm_cls<<<(TT * BB) / 16, 32, 0, stream>>>(l2out, clsSw, cls_b, outf);
  crf_kernel<<<BB, 64, 0, stream>>>(outf, attn, labels, crf_start, crf_end, crf_trans, llh);
  loss_kernel<<<1, 64, 0, stream>>>(llh, outf + (size_t)BB * TT * LL);
}
